// AutoregressivePolicy_45105746542814
// MI455X (gfx1250) — matrix operand order repaired and checked by reading, not compiled, not run on hardware
//
#include <hip/hip_runtime.h>

// ---------------------------------------------------------------------------
// AutoregressivePolicy forward for MI455X (gfx1250), wave32 + WMMA bf16.
//   VOCAB=32000 EMBED=512 HIDDEN=1024 B=2 T=2048  -> ROWS = 4096
// Dominant GEMM: [4096 x 1024] @ [1024 x 32000] via v_wmma_f32_16x16x32_bf16,
// fed by a 3-deep async-to-LDS pipeline (global_load_async_to_lds_b128,
// s_wait_asynccnt), one workgroup barrier per 32-K step.
// ---------------------------------------------------------------------------

#define VOCAB  32000
#define EMBED  512
#define HIDDEN 1024
#define BATCH  2
#define TSEQ   2048
#define ROWS   (BATCH * TSEQ)   // 4096
#define LN_EPS 1e-5f

// LDS tile geometry: 128 rows x 32 halfs, padded to 40 halfs (80B) per row
// so b128 fragment reads are bank-conflict-free across 16 lanes.
#define TPITCH 40                        // halfs per padded row
#define TILEH  (128 * TPITCH)            // halfs per tile (5120 = 10KB)
#define BUFH   (2 * TILEH)               // halfs per stage buffer (A + B) = 20KB
#define BUFB   (BUFH * 2)                // bytes per stage buffer
#define NBUF   3                         // pipeline depth (60KB LDS)

typedef __attribute__((ext_vector_type(16))) __bf16 v16bf;
typedef __attribute__((ext_vector_type(8)))  float  v8f;

__device__ __forceinline__ unsigned short f2bf(float f) {
  unsigned int u = __float_as_uint(f);
  u += 0x7FFFu + ((u >> 16) & 1u);   // round-to-nearest-even
  return (unsigned short)(u >> 16);
}

// Per-lane async 16B copy: global -> LDS, tracked by ASYNCcnt.
__device__ __forceinline__ void async_copy16(unsigned lds_off, const unsigned short* g) {
  asm volatile("global_load_async_to_lds_b128 %0, %1, off"
               :: "v"(lds_off), "v"((unsigned long long)(size_t)g)
               : "memory");
}
__device__ __forceinline__ void wait_async_le4() {
  asm volatile("s_wait_asynccnt 0x4" ::: "memory");
}
__device__ __forceinline__ void wait_async_le0() {
  asm volatile("s_wait_asynccnt 0x0" ::: "memory");
}

// ---------------------------------------------------------------------------
// Weight convert + transpose (LDS-tiled so both sides stay coalesced):
//   W[K][N] fp32  ->  Wt[N][K] bf16.  fp32 read is one-pass -> nontemporal.
// ---------------------------------------------------------------------------
__global__ void wt_bf16_kernel(const float* __restrict__ W,
                               unsigned short* __restrict__ Wt,
                               int K, int N) {
  __shared__ unsigned short tile[32][33];
  const int k0 = blockIdx.y * 32;
  const int n0 = blockIdx.x * 32;
  for (int i = threadIdx.y; i < 32; i += 8) {
    float w = __builtin_nontemporal_load(&W[(long long)(k0 + i) * N + (n0 + threadIdx.x)]);
    tile[i][threadIdx.x] = f2bf(w);
  }
  __syncthreads();
  for (int i = threadIdx.y; i < 32; i += 8) {
    Wt[(long long)(n0 + i) * K + (k0 + threadIdx.x)] = tile[threadIdx.x][i];
  }
}

// ---------------------------------------------------------------------------
// Stage 1: fully parallel coalesced embedding gather (bf16 + fp32 copies).
// ---------------------------------------------------------------------------
__global__ void embed_gather_kernel(const int* __restrict__ tokens,
                                    const float* __restrict__ E,
                                    unsigned short* __restrict__ embBf,
                                    float* __restrict__ embF) {
  long long idx = (long long)blockIdx.x * blockDim.x + threadIdx.x; // ROWS*EMBED
  int d  = (int)(idx & (EMBED - 1));
  int bt = (int)(idx >> 9);              // EMBED = 512 = 2^9
  int tok = tokens[bt];
  float e = E[(long long)tok * EMBED + d];
  embBf[idx] = f2bf(e);
  embF[idx]  = e;
}

// ---------------------------------------------------------------------------
// Stage 2: causal history mean; sequential scan per (b,d) over contiguous fp32.
// ---------------------------------------------------------------------------
__global__ void hist_kernel(const float* __restrict__ embF,
                            unsigned short* __restrict__ histBf) {
  int tid = blockIdx.x * blockDim.x + threadIdx.x;
  if (tid >= BATCH * EMBED) return;
  int b = tid / EMBED;
  int d = tid % EMBED;
  const float* src = embF + (long long)b * TSEQ * EMBED + d;
  unsigned short* dst = histBf + (long long)b * TSEQ * EMBED + d;
  float e0 = src[0];
  dst[0] = f2bf(e0);
  float csum = e0;
#pragma unroll 8
  for (int t = 1; t < TSEQ; ++t) {
    dst[(long long)t * EMBED] = f2bf(csum / (float)t);
    csum += src[(long long)t * EMBED];
  }
}

// ---------------------------------------------------------------------------
// WMMA GEMM with 3-deep async-LDS pipeline:
//   C[M][N] = epilogue( A[M][K](bf16) * Bt[N][K](bf16)^T + bias )
// block = 256 threads (8 waves), tile 128(M) x 128(N); wave = 32x64 (2x4 wmma)
// per 32-K step: 16KB staged, 64 WMMAs per workgroup, ONE barrier per step.
// MODE 0: bias+relu -> bf16   MODE 1: bias+relu -> f32   MODE 2: (..)*0.1 -> f32 NT
// ---------------------------------------------------------------------------
template <int MODE>
__global__ void gemm_wmma_kernel(const unsigned short* __restrict__ A,
                                 const unsigned short* __restrict__ Bt,
                                 const float* __restrict__ bias,
                                 void* __restrict__ outp,
                                 int K, int ldo) {
  __shared__ unsigned short lds[NBUF * BUFH];   // 60KB

  const int tid  = threadIdx.x;
  const int lane = tid & 31;
  const int wave = tid >> 5;          // 0..7
  const int wm   = wave & 3;          // M direction, 0..3 (32 rows each)
  const int wn   = wave >> 2;         // N direction, 0..1 (64 cols each)
  const int lh   = lane >> 4;         // half-wave select
  const int ll   = lane & 15;
  const int rowBlock = blockIdx.y * 128;
  const int colBlock = blockIdx.x * 128;

  v8f acc[2][4];
#pragma unroll
  for (int m = 0; m < 2; ++m)
#pragma unroll
    for (int n = 0; n < 4; ++n)
      acc[m][n] = (v8f){};

  union U16 { uint4 q[2]; v16bf v; };

  // Chunk map: 512 x 16B chunks per tile; each thread copies 2 A + 2 B chunks.
  const int c0 = tid * 2;
  const int r0 = c0 >> 2,       s0 = c0 & 3;
  const int r1 = (c0 + 1) >> 2, s1 = (c0 + 1) & 3;
  const unsigned short* pA0 = A  + (long long)(rowBlock + r0) * K + s0 * 8;
  const unsigned short* pA1 = A  + (long long)(rowBlock + r1) * K + s1 * 8;
  const unsigned short* pB0 = Bt + (long long)(colBlock + r0) * K + s0 * 8;
  const unsigned short* pB1 = Bt + (long long)(colBlock + r1) * K + s1 * 8;
  const unsigned ldsBase = (unsigned)(size_t)(&lds[0]);
  const unsigned aoff0 = ldsBase + r0 * (TPITCH * 2) + s0 * 16;
  const unsigned aoff1 = ldsBase + r1 * (TPITCH * 2) + s1 * 16;
  const unsigned boff0 = aoff0 + TILEH * 2;
  const unsigned boff1 = aoff1 + TILEH * 2;

  // Issues the next 32-K stage into buffer at byte offset `bofB`, advances ptrs.
  auto issue = [&](unsigned bofB) {
    async_copy16(aoff0 + bofB, pA0);
    async_copy16(aoff1 + bofB, pA1);
    async_copy16(boff0 + bofB, pB0);
    async_copy16(boff1 + bofB, pB1);
    pA0 += 32; pA1 += 32; pB0 += 32; pB1 += 32;
  };

  auto computeStep = [&](int baseH) {
    const unsigned short* Abuf = &lds[baseH];
    const unsigned short* Bbuf = Abuf + TILEH;
    U16 a[2], b[4];
#pragma unroll
    for (int m = 0; m < 2; ++m) {
      const unsigned short* p = Abuf + (wm * 32 + m * 16 + ll) * TPITCH + lh * 8;
      a[m].q[0] = *(const uint4*)(p);
      a[m].q[1] = *(const uint4*)(p + 16);
    }
#pragma unroll
    for (int n = 0; n < 4; ++n) {
      const unsigned short* p = Bbuf + (wn * 64 + n * 16 + ll) * TPITCH + lh * 8;
      b[n].q[0] = *(const uint4*)(p);
      b[n].q[1] = *(const uint4*)(p + 16);
    }
#pragma unroll
    for (int m = 0; m < 2; ++m)
#pragma unroll
      for (int n = 0; n < 4; ++n)
        acc[m][n] = __builtin_amdgcn_wmma_f32_16x16x32_bf16(
            false, a[m].v, false, b[n].v, (short)0, acc[m][n], false, false);
  };

  const int nsteps = K >> 5;            // >= 16 for all our shapes

  // Prologue: stage 0 -> buf0, stage 1 -> buf1.
  issue(0);
  issue(BUFB);

  int curH = 0;                         // compute buffer (halfs offset)
  unsigned issB = 2u * BUFB;            // issue buffer (bytes offset)

  // Steady state, branch-free: compute stage s, issue stage s+2.
  for (int s = 0; s < nsteps - 2; ++s) {
    wait_async_le4();                   // stage s resident (s+1 may be in flight)
    __syncthreads();                    // visibility + WAR vs. buffer (s-1)%3
    issue(issB);
    computeStep(curH);
    curH += BUFH; if (curH == NBUF * BUFH) curH = 0;
    issB += BUFB; if (issB == NBUF * (unsigned)BUFB) issB = 0;
  }
  // Tail: stages nsteps-2, nsteps-1 (no more issues).
  wait_async_le4();
  __syncthreads();
  computeStep(curH);
  curH += BUFH; if (curH == NBUF * BUFH) curH = 0;
  wait_async_le0();
  __syncthreads();
  computeStep(curH);

  // Epilogue. C layout: lane -> N = ll, VGPR r -> M = lh*8 + r per 16x16 tile.
#pragma unroll
  for (int m = 0; m < 2; ++m) {
#pragma unroll
    for (int n = 0; n < 4; ++n) {
      const int col = colBlock + wn * 64 + n * 16 + ll;
      const float bv = bias[col];
#pragma unroll
      for (int r = 0; r < 8; ++r) {
        const int row = rowBlock + wm * 32 + m * 16 + lh * 8 + r;
        float v = acc[m][n][r] + bv;
        if (MODE == 0) {
          v = fmaxf(v, 0.0f);
          ((unsigned short*)outp)[(long long)row * ldo + col] = f2bf(v);
        } else if (MODE == 1) {
          v = fmaxf(v, 0.0f);
          ((float*)outp)[(long long)row * ldo + col] = v;
        } else {
          // 524MB logits stream: nontemporal so Wo^T stays L2-resident.
          __builtin_nontemporal_store(v * 0.1f,
              &((float*)outp)[(long long)row * ldo + col]);
        }
      }
    }
  }
}

// ---------------------------------------------------------------------------
// LayerNorm over HIDDEN per row, emits bf16 for the final GEMM's A operand.
// ---------------------------------------------------------------------------
__global__ void layernorm_kernel(const float* __restrict__ X,
                                 const float* __restrict__ ln_w,
                                 const float* __restrict__ ln_b,
                                 unsigned short* __restrict__ outBf) {
  __shared__ float s1[256];
  __shared__ float s2[256];
  const int row = blockIdx.x;
  const float* x = X + (long long)row * HIDDEN;
  float sum = 0.0f, sq = 0.0f;
  for (int i = threadIdx.x; i < HIDDEN; i += 256) {
    float v = x[i];
    sum += v;
    sq  += v * v;
  }
  s1[threadIdx.x] = sum;
  s2[threadIdx.x] = sq;
  __syncthreads();
  for (int s = 128; s > 0; s >>= 1) {
    if (threadIdx.x < s) {
      s1[threadIdx.x] += s1[threadIdx.x + s];
      s2[threadIdx.x] += s2[threadIdx.x + s];
    }
    __syncthreads();
  }
  const float mean = s1[0] * (1.0f / HIDDEN);
  const float var  = s2[0] * (1.0f / HIDDEN) - mean * mean;
  const float inv  = rsqrtf(var + LN_EPS);
  for (int i = threadIdx.x; i < HIDDEN; i += 256) {
    float v = (x[i] - mean) * inv * ln_w[i] + ln_b[i];
    outBf[(long long)row * HIDDEN + i] = f2bf(v);
  }
}

// ---------------------------------------------------------------------------
extern "C" void kernel_launch(void* const* d_in, const int* in_sizes, int n_in,
                              void* d_out, int out_size, void* d_ws, size_t ws_size,
                              hipStream_t stream) {
  const int*   tokens = (const int*)  d_in[0];
  const float* E      = (const float*)d_in[1];
  const float* Wc     = (const float*)d_in[2];
  const float* bc     = (const float*)d_in[3];
  const float* Wh     = (const float*)d_in[4];
  const float* bh     = (const float*)d_in[5];
  const float* Wm     = (const float*)d_in[6];
  const float* bm     = (const float*)d_in[7];
  const float* Wo     = (const float*)d_in[8];
  const float* bo     = (const float*)d_in[9];
  const float* lnw    = (const float*)d_in[10];
  const float* lnb    = (const float*)d_in[11];
  float* out = (float*)d_out;

  char* ws = (char*)d_ws;
  auto carve = [&](size_t bytes) {
    char* p = ws;
    ws += (bytes + 255) & ~(size_t)255;
    return p;
  };
  unsigned short* WcT    = (unsigned short*)carve((size_t)HIDDEN * EMBED * 2);      // 1 MB
  unsigned short* WhT    = (unsigned short*)carve((size_t)HIDDEN * EMBED * 2);      // 1 MB
  unsigned short* WmT    = (unsigned short*)carve((size_t)HIDDEN * 2 * HIDDEN * 2); // 4 MB
  unsigned short* WoT    = (unsigned short*)carve((size_t)VOCAB * HIDDEN * 2);      // 64 MB
  unsigned short* embBf  = (unsigned short*)carve((size_t)ROWS * EMBED * 2);        // 4 MB
  float*          embF   = (float*)         carve((size_t)ROWS * EMBED * 4);        // 8 MB
  unsigned short* histBf = (unsigned short*)carve((size_t)ROWS * EMBED * 2);        // 4 MB
  unsigned short* mergBf = (unsigned short*)carve((size_t)ROWS * 2 * HIDDEN * 2);   // 16 MB
  float*          merg2  = (float*)         carve((size_t)ROWS * HIDDEN * 4);       // 16 MB
  unsigned short* normBf = (unsigned short*)carve((size_t)ROWS * HIDDEN * 2);       // 8 MB
  (void)ws_size; (void)in_sizes; (void)n_in; (void)out_size;

  // 1) weight convert + transpose -> bf16 [N][K]
  {
    dim3 blk(32, 8);
    wt_bf16_kernel<<<dim3(HIDDEN / 32, EMBED / 32),      blk, 0, stream>>>(Wc, WcT, EMBED,      HIDDEN);
    wt_bf16_kernel<<<dim3(HIDDEN / 32, EMBED / 32),      blk, 0, stream>>>(Wh, WhT, EMBED,      HIDDEN);
    wt_bf16_kernel<<<dim3(HIDDEN / 32, 2 * HIDDEN / 32), blk, 0, stream>>>(Wm, WmT, 2 * HIDDEN, HIDDEN);
    wt_bf16_kernel<<<dim3(VOCAB / 32, HIDDEN / 32),      blk, 0, stream>>>(Wo, WoT, HIDDEN,     VOCAB);
  }

  // 2) embeddings (parallel gather) + causal history mean (light scan)
  embed_gather_kernel<<<(ROWS * EMBED) / 256, 256, 0, stream>>>(tokens, E, embBf, embF);
  hist_kernel<<<(BATCH * EMBED + 255) / 256, 256, 0, stream>>>(embF, histBf);

  // 3) encoders: relu(emb@Wc+bc), relu(hist@Wh+bh) -> concat buffer [ROWS][2H]
  {
    dim3 grid(HIDDEN / 128, ROWS / 128);
    gemm_wmma_kernel<0><<<grid, 256, 0, stream>>>(embBf,  WcT, bc, mergBf,          EMBED, 2 * HIDDEN);
    gemm_wmma_kernel<0><<<grid, 256, 0, stream>>>(histBf, WhT, bh, mergBf + HIDDEN, EMBED, 2 * HIDDEN);
  }

  // 4) merge: relu(concat @ Wm + bm) -> fp32 for layernorm
  {
    dim3 grid(HIDDEN / 128, ROWS / 128);
    gemm_wmma_kernel<1><<<grid, 256, 0, stream>>>(mergBf, WmT, bm, merg2, 2 * HIDDEN, HIDDEN);
  }

  // 5) layernorm -> bf16
  layernorm_kernel<<<ROWS, 256, 0, stream>>>(merg2, lnw, lnb, normBf);

  // 6) logits: (normed @ Wo + bo) * 0.1 -> fp32 out   [dominant GEMM]
  {
    dim3 grid(VOCAB / 128, ROWS / 128);
    gemm_wmma_kernel<2><<<grid, 256, 0, stream>>>(normBf, WoT, bo, out, HIDDEN, VOCAB);
  }
}
